// LobsterDynamics_83708912599088
// MI455X (gfx1250) — matrix operand order repaired and checked by reading, not compiled, not run on hardware
//
#include <hip/hip_runtime.h>

typedef __attribute__((ext_vector_type(16))) _Float16 v16h;
typedef __attribute__((ext_vector_type(4)))  _Float16 v4h;
typedef __attribute__((ext_vector_type(2)))  _Float16 v2h;
typedef __attribute__((ext_vector_type(8)))  float    v8f;

#define HID 64
#define NLAYER 4

__device__ __forceinline__ float silu_f(float x) { return x / (1.0f + __expf(-x)); }

// Load a 16x32 f16 A-fragment (v16h per lane) from a row-major LDS tile.
// 'p' must already be: tile + (lane&15)*row_stride + kbase + 8*(lane>>4).
// Per ISA 16-bit A layout: VGPR v<4 holds K=2v,2v+1 ; v>=4 holds K=16+2(v-4)+{0,1},
// with lanes 16-31 shifted by +8 in K (folded into 'p').
__device__ __forceinline__ v16h lds_afrag(const _Float16* p) {
    v16h a;
#pragma unroll
    for (int v = 0; v < 8; ++v) {
        const int k = (v < 4) ? (2 * v) : (16 + 2 * (v - 4));
        v2h t = *(const v2h*)(p + k);
        a[2 * v]     = t[0];
        a[2 * v + 1] = t[1];
    }
    return a;
}

__device__ __forceinline__ v8f wmma16(v16h a, v16h b, v8f c) {
    return __builtin_amdgcn_wmma_f32_16x16x32_f16(false, a, false, b, (short)0, c,
                                                  false, false);
}

// Pre-swizzle an f32 weight matrix W[din][64] into per-lane B-fragment order (f16).
// Fragment f = kt*4 + nt. Element (lane, half j):
//   B[k][n] with k = kt*32 + 8*(lane>>4) + j + 8*(j>>3), n = nt*16 + (lane&15).
__global__ void prep_w(const float* __restrict__ W, _Float16* __restrict__ dst, int din) {
    const int idx = blockIdx.x * 256 + threadIdx.x;
    if (idx >= din * 64) return;
    const int j    = idx & 15;
    const int lane = (idx >> 4) & 31;
    const int f    = idx >> 9;
    const int kt = f >> 2, nt = f & 3;
    const int k = kt * 32 + 8 * (lane >> 4) + j + 8 * (j >> 3);
    const int n = nt * 16 + (lane & 15);
    dst[idx] = (_Float16)W[k * 64 + n];
}

// h[i][j] = x[i]*W[0][j] + t*W[1][j] + b[j]
__global__ void embed_node(const float* __restrict__ x, const float* __restrict__ t,
                           const float* __restrict__ W, const float* __restrict__ b,
                           float* __restrict__ h, int nN) {
    const int i = blockIdx.x * 256 + threadIdx.x;
    if (i >= nN * HID) return;
    const int node = i >> 6, j = i & 63;
    h[i] = x[node] * W[j] + t[0] * W[HID + j] + b[j];
}

// e[i][j] = a0*W[0][j] + a1*W[1][j] + t*W[2][j] + b[j]
__global__ void embed_edge(const float* __restrict__ ea, const float* __restrict__ t,
                           const float* __restrict__ W, const float* __restrict__ b,
                           float* __restrict__ e, int nE) {
    const int i = blockIdx.x * 256 + threadIdx.x;
    if (i >= nE * HID) return;
    const int ed = i >> 6, j = i & 63;
    e[i] = ea[ed * 2] * W[j] + ea[ed * 2 + 1] * W[HID + j] + t[0] * W[2 * HID + j] + b[j];
}

// Edge model: m = silu(W2 @ silu(W1 @ [h[row]|h[col]|e] + b1) + b2); e += m;
// agg[row] += m (scatter via f32 atomics; /NORM applied at node staging).
// One wave = 16 edges. nE must be a multiple of 16 (it is: 800000).
__global__ void __launch_bounds__(256) edge_mlp(
    const float* __restrict__ hbuf, float* __restrict__ ebuf, float* __restrict__ agg,
    const int* __restrict__ eidx,
    const _Float16* __restrict__ w1, const float* __restrict__ b1,
    const _Float16* __restrict__ w2, const float* __restrict__ b2,
    int nE)
{
    constexpr int TS = 200;                 // row stride (halves), padded vs 192
    __shared__ _Float16 sh[8 * 16 * TS];    // 8 waves * 16x200 f16 = 51200 B
    const int lane = threadIdx.x & 31;
    const int wv   = threadIdx.x >> 5;
    const int tileId = blockIdx.x * 8 + wv;
    const long long eb = (long long)tileId * 16;
    if (eb >= nE) return;
    _Float16* tile = sh + wv * (16 * TS);
    const int* __restrict__ rowp = eidx;
    const int* __restrict__ colp = eidx + nE;
    const int lm = lane & 15, hi = lane >> 4;

    // ---- stage [h[row] | h[col] | e] as a 16 x 192 f16 tile ----
#pragma unroll
    for (int it = 0; it < 8; ++it) {
        const int i = lane + it * 32;
        const int m = i >> 4, q = i & 15;
        const long long eg = eb + m;
        const int r = rowp[eg];
        const int c = colp[eg];
        const float4 f0 = *(const float4*)(hbuf + (long long)r * HID + q * 4);
        const float4 f1 = *(const float4*)(hbuf + (long long)c * HID + q * 4);
        const float4 f2 = *(const float4*)(ebuf + eg * HID + q * 4);
        _Float16* rp = tile + m * TS + q * 4;
        v4h a0 = {(_Float16)f0.x, (_Float16)f0.y, (_Float16)f0.z, (_Float16)f0.w};
        v4h a1 = {(_Float16)f1.x, (_Float16)f1.y, (_Float16)f1.z, (_Float16)f1.w};
        v4h a2 = {(_Float16)f2.x, (_Float16)f2.y, (_Float16)f2.z, (_Float16)f2.w};
        *(v4h*)(rp)       = a0;
        *(v4h*)(rp + 64)  = a1;
        *(v4h*)(rp + 128) = a2;
    }
    // wave-local LDS RAW: wait for our own stores (lockstep covers cross-lane)
    asm volatile("s_wait_dscnt 0" ::: "memory");

    const _Float16* arow = tile + lm * TS + 8 * hi;
    v16h A[6];
#pragma unroll
    for (int kt = 0; kt < 6; ++kt) A[kt] = lds_afrag(arow + kt * 32);

    // e1: 192 -> 64
    v8f acc[4];
#pragma unroll
    for (int nt = 0; nt < 4; ++nt) {
        const float bv = b1[nt * 16 + lm];
        v8f c = {bv, bv, bv, bv, bv, bv, bv, bv};
#pragma unroll
        for (int kt = 0; kt < 6; ++kt) {
            v16h B = *(const v16h*)(w1 + (size_t)((kt * 4 + nt) * 32 + lane) * 16);
            c = wmma16(A[kt], B, c);
        }
        acc[nt] = c;
    }

    // silu -> 16x64 activation tile in LDS (cols 0..63; A regs already loaded)
#pragma unroll
    for (int nt = 0; nt < 4; ++nt)
#pragma unroll
        for (int r = 0; r < 8; ++r)
            tile[(r + 8 * hi) * TS + nt * 16 + lm] = (_Float16)silu_f(acc[nt][r]);
    asm volatile("s_wait_dscnt 0" ::: "memory");

    v16h A2[2];
    A2[0] = lds_afrag(arow);
    A2[1] = lds_afrag(arow + 32);

    int rm[8];
#pragma unroll
    for (int r = 0; r < 8; ++r) rm[r] = rowp[eb + r + 8 * hi];

    // e2: 64 -> 64, silu, fused writeback
#pragma unroll
    for (int nt = 0; nt < 4; ++nt) {
        const float bv = b2[nt * 16 + lm];
        v8f c = {bv, bv, bv, bv, bv, bv, bv, bv};
        v16h B0 = *(const v16h*)(w2 + (size_t)(nt * 32 + lane) * 16);
        c = wmma16(A2[0], B0, c);
        v16h B1 = *(const v16h*)(w2 + (size_t)((4 + nt) * 32 + lane) * 16);
        c = wmma16(A2[1], B1, c);
        const int n = nt * 16 + lm;
#pragma unroll
        for (int r = 0; r < 8; ++r) {
            const float mv = silu_f(c[r]);
            const long long eo = (eb + r + 8 * hi) * HID + n;
            ebuf[eo] += mv;
            atomicAdd(agg + (long long)rm[r] * HID + n, mv);
        }
    }
}

// Node model: h += W2 @ silu(W1 @ [h | agg/100] + b1) + b2. One wave = 16 nodes.
__global__ void __launch_bounds__(256) node_mlp(
    float* __restrict__ hbuf, const float* __restrict__ agg,
    const _Float16* __restrict__ w1, const float* __restrict__ b1,
    const _Float16* __restrict__ w2, const float* __restrict__ b2,
    int nN)
{
    constexpr int TS = 136;                 // 128 + pad
    __shared__ _Float16 sh[8 * 16 * TS];    // 34816 B
    const int lane = threadIdx.x & 31;
    const int wv   = threadIdx.x >> 5;
    const int tileId = blockIdx.x * 8 + wv;
    const long long nb = (long long)tileId * 16;
    if (nb >= nN) return;
    _Float16* tile = sh + wv * (16 * TS);
    const int lm = lane & 15, hi = lane >> 4;

#pragma unroll
    for (int it = 0; it < 8; ++it) {
        const int i = lane + it * 32;
        const int m = i >> 4, q = i & 15;
        const long long ng = nb + m;
        const float4 f0 = *(const float4*)(hbuf + ng * HID + q * 4);
        const float4 f1 = *(const float4*)(agg  + ng * HID + q * 4);
        _Float16* rp = tile + m * TS + q * 4;
        v4h a0 = {(_Float16)f0.x, (_Float16)f0.y, (_Float16)f0.z, (_Float16)f0.w};
        v4h a1 = {(_Float16)(f1.x * 0.01f), (_Float16)(f1.y * 0.01f),
                  (_Float16)(f1.z * 0.01f), (_Float16)(f1.w * 0.01f)};
        *(v4h*)(rp)      = a0;
        *(v4h*)(rp + 64) = a1;
    }
    asm volatile("s_wait_dscnt 0" ::: "memory");

    const _Float16* arow = tile + lm * TS + 8 * hi;
    v16h A[4];
#pragma unroll
    for (int kt = 0; kt < 4; ++kt) A[kt] = lds_afrag(arow + kt * 32);

    v8f acc[4];
#pragma unroll
    for (int nt = 0; nt < 4; ++nt) {
        const float bv = b1[nt * 16 + lm];
        v8f c = {bv, bv, bv, bv, bv, bv, bv, bv};
#pragma unroll
        for (int kt = 0; kt < 4; ++kt) {
            v16h B = *(const v16h*)(w1 + (size_t)((kt * 4 + nt) * 32 + lane) * 16);
            c = wmma16(A[kt], B, c);
        }
        acc[nt] = c;
    }

#pragma unroll
    for (int nt = 0; nt < 4; ++nt)
#pragma unroll
        for (int r = 0; r < 8; ++r)
            tile[(r + 8 * hi) * TS + nt * 16 + lm] = (_Float16)silu_f(acc[nt][r]);
    asm volatile("s_wait_dscnt 0" ::: "memory");

    v16h A2[2];
    A2[0] = lds_afrag(arow);
    A2[1] = lds_afrag(arow + 32);

#pragma unroll
    for (int nt = 0; nt < 4; ++nt) {
        const float bv = b2[nt * 16 + lm];
        v8f c = {bv, bv, bv, bv, bv, bv, bv, bv};
        v16h B0 = *(const v16h*)(w2 + (size_t)(nt * 32 + lane) * 16);
        c = wmma16(A2[0], B0, c);
        v16h B1 = *(const v16h*)(w2 + (size_t)((4 + nt) * 32 + lane) * 16);
        c = wmma16(A2[1], B1, c);
        const int n = nt * 16 + lm;
#pragma unroll
        for (int r = 0; r < 8; ++r)
            hbuf[(nb + r + 8 * hi) * HID + n] += c[r];   // residual, no activation
    }
}

// out_x[i] = h[i,:] . W[:,0] + b
__global__ void out_node_k(const float* __restrict__ h, const float* __restrict__ W,
                           const float* __restrict__ b, float* __restrict__ out, int nN) {
    const int i = blockIdx.x * 256 + threadIdx.x;
    if (i >= nN) return;
    float acc = b[0];
#pragma unroll
    for (int k = 0; k < HID; ++k) acc += h[(long long)i * HID + k] * W[k];
    out[i] = acc;
}

// out_e[e][j] = e[e,:] . W[:,j] + b[j], dout = 2. 128 edges per block via LDS.
__global__ void __launch_bounds__(256) out_edge_k(
    const float* __restrict__ e, const float* __restrict__ W,
    const float* __restrict__ b, float* __restrict__ out, int nE) {
    __shared__ float s[128 * 65];
    const long long base = (long long)blockIdx.x * 128;
    for (int i = threadIdx.x; i < 128 * HID; i += 256) {
        const int m = i >> 6, k = i & 63;
        s[m * 65 + k] = e[(base + m) * HID + k];
    }
    __syncthreads();
    const int m = threadIdx.x >> 1, j = threadIdx.x & 1;
    float acc = b[j];
#pragma unroll
    for (int k = 0; k < HID; ++k) acc += s[m * 65 + k] * W[k * 2 + j];
    out[(base + m) * 2 + j] = acc;
}

extern "C" void kernel_launch(void* const* d_in, const int* in_sizes, int n_in,
                              void* d_out, int out_size, void* d_ws, size_t ws_size,
                              hipStream_t stream)
{
    const float *x, *ea, *t;
    const int* eidx;
    const float *Wen, *ben, *Wee, *bee, *Won, *bon, *Woe, *boe;
    const float *We1[NLAYER], *be1[NLAYER], *We2[NLAYER], *be2[NLAYER];
    const float *Wn1[NLAYER], *bn1[NLAYER], *Wn2[NLAYER], *bn2[NLAYER];
    int nN, nE;

    if (in_sizes[0] > 1) {
        // setup_inputs() insertion order
        x    = (const float*)d_in[0];
        ea   = (const float*)d_in[1];
        t    = (const float*)d_in[2];
        eidx = (const int*)  d_in[3];
        nN = in_sizes[0];
        nE = in_sizes[3] / 2;
        Wen = (const float*)d_in[7];  ben = (const float*)d_in[8];
        Wee = (const float*)d_in[9];  bee = (const float*)d_in[10];
        Won = (const float*)d_in[11]; bon = (const float*)d_in[12];
        Woe = (const float*)d_in[13]; boe = (const float*)d_in[14];
        for (int l = 0; l < NLAYER; ++l) {
            const int b0 = 15 + 8 * l;
            We1[l] = (const float*)d_in[b0 + 0]; be1[l] = (const float*)d_in[b0 + 1];
            We2[l] = (const float*)d_in[b0 + 2]; be2[l] = (const float*)d_in[b0 + 3];
            Wn1[l] = (const float*)d_in[b0 + 4]; bn1[l] = (const float*)d_in[b0 + 5];
            Wn2[l] = (const float*)d_in[b0 + 6]; bn2[l] = (const float*)d_in[b0 + 7];
        }
    } else {
        // jax pytree (alphabetical) order: batch_size, edge_attr, edge_index,
        // edge_mask, node_mask, params{emb_edge, emb_node, layers[e1,e2,n1,n2]x4,
        // out_edge, out_node}, t, x
        ea   = (const float*)d_in[1];
        eidx = (const int*)  d_in[2];
        nE = in_sizes[2] / 2;
        nN = in_sizes[4];
        Wee = (const float*)d_in[5]; bee = (const float*)d_in[6];
        Wen = (const float*)d_in[7]; ben = (const float*)d_in[8];
        for (int l = 0; l < NLAYER; ++l) {
            const int b0 = 9 + 8 * l;
            We1[l] = (const float*)d_in[b0 + 0]; be1[l] = (const float*)d_in[b0 + 1];
            We2[l] = (const float*)d_in[b0 + 2]; be2[l] = (const float*)d_in[b0 + 3];
            Wn1[l] = (const float*)d_in[b0 + 4]; bn1[l] = (const float*)d_in[b0 + 5];
            Wn2[l] = (const float*)d_in[b0 + 6]; bn2[l] = (const float*)d_in[b0 + 7];
        }
        Woe = (const float*)d_in[41]; boe = (const float*)d_in[42];
        Won = (const float*)d_in[43]; bon = (const float*)d_in[44];
        t   = (const float*)d_in[45];
        x   = (const float*)d_in[46];
    }

    // ---- workspace layout ----
    const size_t hBytes   = (size_t)nN * HID * sizeof(float);   // 12.8 MB
    const size_t eBytes   = (size_t)nE * HID * sizeof(float);   // 204.8 MB
    const size_t aggBytes = hBytes;
    const size_t wfHalves = (size_t)NLAYER * (192 + 64 + 128 + 64) * 64;
    if (ws_size < hBytes + eBytes + aggBytes + wfHalves * 2) return;

    char* ws = (char*)d_ws;
    float*     hb = (float*)ws;
    float*     eb = (float*)(ws + hBytes);
    float*     ag = (float*)(ws + hBytes + eBytes);
    _Float16*  wf = (_Float16*)(ws + hBytes + eBytes + aggBytes);

    _Float16 *we1[NLAYER], *we2[NLAYER], *wn1[NLAYER], *wn2[NLAYER];
    {
        size_t off = 0;
        for (int l = 0; l < NLAYER; ++l) {
            we1[l] = wf + off; off += 192 * 64;
            we2[l] = wf + off; off += 64 * 64;
            wn1[l] = wf + off; off += 128 * 64;
            wn2[l] = wf + off; off += 64 * 64;
        }
    }

    // ---- 1. pre-swizzle weights to f16 B-fragment order ----
    for (int l = 0; l < NLAYER; ++l) {
        prep_w<<<(192 * 64 + 255) / 256, 256, 0, stream>>>(We1[l], we1[l], 192);
        prep_w<<<(64 * 64 + 255) / 256, 256, 0, stream>>>(We2[l], we2[l], 64);
        prep_w<<<(128 * 64 + 255) / 256, 256, 0, stream>>>(Wn1[l], wn1[l], 128);
        prep_w<<<(64 * 64 + 255) / 256, 256, 0, stream>>>(Wn2[l], wn2[l], 64);
    }

    // ---- 2. embeddings ----
    embed_node<<<(nN * HID + 255) / 256, 256, 0, stream>>>(x, t, Wen, ben, hb, nN);
    embed_edge<<<(nE * HID + 255) / 256, 256, 0, stream>>>(ea, t, Wee, bee, eb, nE);

    // ---- 3. GNN layers ----
    const int edgeBlocks = (nE / 16 + 7) / 8;   // 6250
    const int nodeBlocks = (nN / 16 + 7) / 8;   // 391
    for (int l = 0; l < NLAYER; ++l) {
        hipMemsetAsync(ag, 0, aggBytes, stream);
        edge_mlp<<<edgeBlocks, 256, 0, stream>>>(hb, eb, ag, eidx,
                                                 we1[l], be1[l], we2[l], be2[l], nE);
        node_mlp<<<nodeBlocks, 256, 0, stream>>>(hb, ag,
                                                 wn1[l], bn1[l], wn2[l], bn2[l], nN);
    }

    // ---- 4. output heads ----
    float* out_x = (float*)d_out;
    float* out_e = out_x + nN;
    out_node_k<<<(nN + 255) / 256, 256, 0, stream>>>(hb, Won, bon, out_x, nN);
    out_edge_k<<<nE / 128, 256, 0, stream>>>(eb, Woe, boe, out_e, nE);
}
